// ResVertixRefineShapenet_2259152797813
// MI455X (gfx1250) — matrix operand order repaired and checked by reading, not compiled, not run on hardware
//
#include <hip/hip_runtime.h>

typedef __attribute__((ext_vector_type(16))) __bf16 v16bf;
typedef __attribute__((ext_vector_type(8)))  __bf16 v8bf;
typedef __attribute__((ext_vector_type(8)))  float  v8f;

#define NVERT    8192
#define NFEAT    128
#define ALIGN_CH 3840
#define IN0      259
#define IN0P     288   // 259 padded up to multiple of 32

// ---------------------------------------------------------------------------
// helpers
// ---------------------------------------------------------------------------
static __device__ __forceinline__ unsigned short f32_to_bf16(float f) {
    unsigned int u = __float_as_uint(f);
    unsigned int lsb = (u >> 16) & 1u;
    u += 0x7FFFu + lsb;                 // round-to-nearest-even
    return (unsigned short)(u >> 16);
}

// ---------------------------------------------------------------------------
// adj f32 -> bf16 (0/1 values: conversion is exact), vectorized
// ---------------------------------------------------------------------------
__global__ void cvt_bf16_vec_kernel(const float* __restrict__ src,
                                    unsigned short* __restrict__ dst, size_t n) {
    size_t i = ((size_t)blockIdx.x * blockDim.x + threadIdx.x) * 4;
    size_t stride = (size_t)gridDim.x * blockDim.x * 4;
    for (; i < n; i += stride) {
        float4 f = *(const float4*)(src + i);
        ushort4 r;
        r.x = f32_to_bf16(f.x); r.y = f32_to_bf16(f.y);
        r.z = f32_to_bf16(f.z); r.w = f32_to_bf16(f.w);
        *(ushort4*)(dst + i) = r;
    }
}

// ---------------------------------------------------------------------------
// weight pad + convert: f32 (K,N) -> bf16 (Kp,Np) zero-filled
// ---------------------------------------------------------------------------
__global__ void cvt_pad_kernel(const float* __restrict__ src,
                               unsigned short* __restrict__ dst,
                               int K, int N, int Kp, int Np) {
    int tot = Kp * Np;
    for (int i = blockIdx.x * blockDim.x + threadIdx.x; i < tot;
         i += gridDim.x * blockDim.x) {
        int k = i / Np, n = i % Np;
        float v = (k < K && n < N) ? src[(size_t)k * N + n] : 0.0f;
        dst[i] = f32_to_bf16(v);
    }
}

// ---------------------------------------------------------------------------
// vertex_align (faithful: truncated integer weights, x indexes dim -2,
// y indexes dim -1), output directly in bf16 row-major (8192 x 3840)
// ---------------------------------------------------------------------------
__global__ void vertex_align_kernel(const float* __restrict__ f0,
                                    const float* __restrict__ f1,
                                    const float* __restrict__ f2,
                                    const float* __restrict__ f3,
                                    const float* __restrict__ pos,
                                    unsigned short* __restrict__ alignedb) {
    int v = blockIdx.x;
    float px = pos[v * 3 + 0], py = pos[v * 3 + 1], pz = pos[v * 3 + 2];
    float h = fminf(fmaxf(248.0f * (py / pz)    + 111.5f, 0.0f), 223.0f);
    float w = fminf(fmaxf(248.0f * (px / (-pz)) + 111.5f, 0.0f), 223.0f);

    const float* feats[4] = {f0, f1, f2, f3};
    const int sdim[4] = {56, 28, 14, 7};
    const int coff[4] = {0, 256, 768, 1792};

    for (int c = threadIdx.x; c < ALIGN_CH; c += blockDim.x) {
        int fi = (c < 256) ? 0 : (c < 768) ? 1 : (c < 1792) ? 2 : 3;
        int cl = c - coff[fi];
        int s  = sdim[fi];
        float x = w * ((float)s / 224.0f);
        float y = h * ((float)s / 224.0f);
        int x1 = (int)floorf(x);
        int x2 = min((int)ceilf(x), s - 1);
        int y1 = (int)floorf(y);
        int y2 = min((int)ceilf(y), s - 1);
        int xi = (int)x, yi = (int)y;
        float w11 = (float)((x2 - xi) * (y2 - yi));
        float w12 = (float)((x2 - xi) * (yi - y1));
        float w21 = (float)((xi - x1) * (y2 - yi));
        float w22 = (float)((xi - x1) * (yi - y1));
        const float* fb = feats[fi] + (size_t)cl * s * s;
        float q11 = fb[x1 * s + y1];
        float q12 = fb[x1 * s + y2];
        float q21 = fb[x2 * s + y1];
        float q22 = fb[x2 * s + y2];
        float val = w11 * q11 + w21 * q21 + w12 * q12 + w22 * q22;
        alignedb[(size_t)v * ALIGN_CH + c] = f32_to_bf16(val);
    }
}

// ---------------------------------------------------------------------------
// concat -> x0 bf16 padded: [vfeat(128) | pos(3) | projected(128) | zeros(29)]
// ---------------------------------------------------------------------------
__global__ void concat_kernel(const float* __restrict__ vfeat,
                              const float* __restrict__ pos,
                              const float* __restrict__ P,
                              unsigned short* __restrict__ x0b) {
    int v = blockIdx.x;
    for (int c = threadIdx.x; c < IN0P; c += blockDim.x) {
        float val;
        if (c < 128)       val = vfeat[(size_t)v * 128 + c];
        else if (c < 131)  val = pos[v * 3 + (c - 128)];
        else if (c < 259)  val = P[(size_t)v * 128 + (c - 131)];
        else               val = 0.0f;
        x0b[(size_t)v * IN0P + c] = f32_to_bf16(val);
    }
}

// ---------------------------------------------------------------------------
// elementwise: of = a + b  (a may alias of), also write bf16 copy
// ---------------------------------------------------------------------------
__global__ void add_dual_kernel(const float* __restrict__ a,
                                const float* __restrict__ b,
                                float* __restrict__ of,
                                unsigned short* __restrict__ ob, size_t n) {
    size_t i = (size_t)blockIdx.x * blockDim.x + threadIdx.x;
    if (i < n) {
        float v = a[i] + b[i];
        of[i] = v;
        ob[i] = f32_to_bf16(v);
    }
}

// ---------------------------------------------------------------------------
// final: out[0:8192*3] = pos + tanh(D[:, :3]);  out[8192*3:] = X (8192x128)
// ---------------------------------------------------------------------------
__global__ void final_kernel(const float* __restrict__ pos,
                             const float* __restrict__ D,   // 8192 x 16
                             const float* __restrict__ X,   // 8192 x 128
                             float* __restrict__ out) {
    size_t i = (size_t)blockIdx.x * blockDim.x + threadIdx.x;
    if (i < (size_t)NVERT * 3) {
        int v = (int)(i / 3), d = (int)(i % 3);
        out[i] = pos[i] + tanhf(D[(size_t)v * 16 + d]);
    }
    if (i < (size_t)NVERT * NFEAT) {
        out[(size_t)NVERT * 3 + i] = X[i];
    }
}

// ---------------------------------------------------------------------------
// WMMA bf16 GEMM:  Cf(f32, MxN) = A(bf16, MxK rowmaj) @ B(bf16, KxN rowmaj)
//   optional: += Aux, ReLU, dual bf16 store.
//
// 256 threads = 8 waves. Each wave: 16 rows x (NB*16) cols -> NB v8f accums.
//
// B-tile pipeline (per 32-K step):
//   1) global_load_async_to_lds_b128 copies the raw 32xNtile bf16 tile into
//      ldsRaw[buf] (ASYNCcnt; GV addressing; per-lane 16B chunks), issued one
//      iteration ahead so the copy overlaps the WMMA phase (double-buffered).
//   2) an LDS->LDS swizzle pass rewrites it into a contiguous-k (k = 16h + s) 16-bit
//      B-fragment layout (lane = 16*(k>=16)+n; elem pair j = K 2j,2j+1) so
//      each lane's fragment is 32 contiguous bytes (2x ds_load_b128).
// A fragments in the same contiguous-k order (k = 16*hk + s) straight from global (2x b128).
// All NB B fragments are batch-loaded before the WMMA chain so the 8 WMMAs
// issue back-to-back instead of stalling on DScnt each.
// M % 128 == 0, K % 32 == 0.
// ---------------------------------------------------------------------------
template <int NB>
__global__ __launch_bounds__(256)
void gemm_bf16_wmma(const unsigned short* __restrict__ A,
                    const unsigned short* __restrict__ B,
                    float* __restrict__ Cf,
                    unsigned short* __restrict__ Cb,
                    const float* __restrict__ Aux,
                    int M, int N, int K, int relu) {
    __shared__ unsigned short ldsRaw[2][32 * 128];   // raw row-major tiles
    __shared__ unsigned short ldsFrag[32 * 128];     // fragment-swizzled tile

    const int t    = threadIdx.x;
    const int lane = t & 31;
    const int wave = t >> 5;
    const int row0 = blockIdx.x * 128 + wave * 16;
    const int col0 = blockIdx.y * 128;
    const int Ntile = NB * 16;

    v8f acc[NB];
#pragma unroll
    for (int i = 0; i < NB; ++i)
#pragma unroll
        for (int j = 0; j < 8; ++j) acc[i][j] = 0.0f;

    const int m  = lane & 15;
    const int hk = lane >> 4;
    const unsigned short* Arow = A + (size_t)(row0 + m) * K;

    // async copy of one raw B tile (32 x Ntile bf16) into ldsRaw[rb]
    auto asyncStage = [&](int rb, int k0) {
        const int chunksPerRow = Ntile / 8;          // 16B chunks per row
        const int chunks = 32 * chunksPerRow;
        for (int c = t; c < chunks; c += 256) {
            int row = c / chunksPerRow;
            int cc  = c % chunksPerRow;
            const unsigned short* g =
                B + (size_t)(k0 + row) * N + col0 + cc * 8;
            unsigned lds_off =
                (unsigned)(size_t)&ldsRaw[rb][row * Ntile + cc * 8];
            asm volatile("global_load_async_to_lds_b128 %0, %1, off"
                         :: "v"(lds_off), "v"(g)
                         : "memory");
        }
    };

    // LDS->LDS rewrite into WMMA B-fragment order
    auto swizzleB = [&](int rb) {
        for (int p = t; p < 16 * Ntile; p += 256) {
            int n  = p % Ntile;
            int kp = p / Ntile;
            int k  = kp * 2;
            unsigned int b0 = ldsRaw[rb][k * Ntile + n];
            unsigned int b1 = ldsRaw[rb][(k + 1) * Ntile + n];
            int nb = n >> 4, nn = n & 15;
            int kh = k >> 4, j = (k & 15) >> 1;
            int idx = ((nb * 32 + kh * 16 + nn) << 4) + j * 2;
            *(unsigned int*)(ldsFrag + idx) = b0 | (b1 << 16);
        }
    };

    asyncStage(0, 0);
    int buf = 0;

    for (int k0 = 0; k0 < K; k0 += 32) {
        // my chunks of ldsRaw[buf] have landed; barrier covers everyone's
        asm volatile("s_wait_asynccnt 0x0" ::: "memory");
        __syncthreads();

        swizzleB(buf);
        if (k0 + 32 < K) asyncStage(buf ^ 1, k0 + 32);  // overlaps WMMA phase
        __syncthreads();

        __builtin_prefetch(Arow + k0 + 128, 0, 3);      // global_prefetch_b8

        v8bf alo = *(const v8bf*)(Arow + k0 + hk * 16);
        v8bf ahi = *(const v8bf*)(Arow + k0 + hk * 16 + 8);
        v16bf afrag = __builtin_shufflevector(
            alo, ahi, 0, 1, 2, 3, 4, 5, 6, 7, 8, 9, 10, 11, 12, 13, 14, 15);

        v16bf bfrag[NB];
#pragma unroll
        for (int nb = 0; nb < NB; ++nb)
            bfrag[nb] = *(const v16bf*)(ldsFrag + ((nb * 32 + lane) << 4));

#pragma unroll
        for (int nb = 0; nb < NB; ++nb)
            acc[nb] = __builtin_amdgcn_wmma_f32_16x16x32_bf16(
                false, afrag, false, bfrag[nb], (short)0, acc[nb], false, false);

        buf ^= 1;
    }

    // C layout: VGPR r, lanes 0-15 -> M=r, lanes 16-31 -> M=8+r, N = lane&15
#pragma unroll
    for (int nb = 0; nb < NB; ++nb) {
        int n = col0 + nb * 16 + (lane & 15);
#pragma unroll
        for (int r = 0; r < 8; ++r) {
            int mm = row0 + r + hk * 8;
            size_t o = (size_t)mm * N + n;
            float v = acc[nb][r];
            if (Aux)  v += Aux[o];
            if (relu) v = v > 0.0f ? v : 0.0f;
            Cf[o] = v;
            if (Cb) Cb[o] = f32_to_bf16(v);
        }
    }
}

// ---------------------------------------------------------------------------
// launcher
// ---------------------------------------------------------------------------
extern "C" void kernel_launch(void* const* d_in, const int* in_sizes, int n_in,
                              void* d_out, int out_size, void* d_ws, size_t ws_size,
                              hipStream_t stream) {
    (void)in_sizes; (void)n_in; (void)out_size; (void)ws_size;

    const float* f0    = (const float*)d_in[0];
    const float* f1    = (const float*)d_in[1];
    const float* f2    = (const float*)d_in[2];
    const float* f3    = (const float*)d_in[3];
    const float* adj   = (const float*)d_in[4];
    const float* pos   = (const float*)d_in[5];
    const float* vfeat = (const float*)d_in[6];
    const float* lin_w = (const float*)d_in[7];
    const float* W[15];
    for (int i = 0; i < 15; ++i) W[i] = (const float*)d_in[8 + i];
    // W[0]=rg0_c0_w0 W[1]=rg0_c0_w1 W[2]=rg0_c1_w0 W[3]=rg0_c1_w1 W[4]=rg0_proj
    // W[5..8]=rg1(c0w0,c0w1,c1w0,c1w1)  W[9..12]=rg2(...)  W[13]=gc_w0 W[14]=gc_w1

    char* ws = (char*)d_ws;
    size_t off = 0;
    auto ALLOC = [&](size_t bytes) -> char* {
        off = (off + 255) & ~(size_t)255;
        char* p = ws + off;
        off += bytes;
        return p;
    };

    unsigned short* adjb     = (unsigned short*)ALLOC((size_t)NVERT * NVERT * 2);
    unsigned short* alignedb = (unsigned short*)ALLOC((size_t)NVERT * ALIGN_CH * 2);
    unsigned short* x0b      = (unsigned short*)ALLOC((size_t)NVERT * IN0P * 2);
    unsigned short* linb     = (unsigned short*)ALLOC((size_t)ALIGN_CH * NFEAT * 2);
    unsigned short* wpad0    = (unsigned short*)ALLOC((size_t)IN0P * NFEAT * 2);
    unsigned short* wpad1    = (unsigned short*)ALLOC((size_t)IN0P * NFEAT * 2);
    unsigned short* wproj    = (unsigned short*)ALLOC((size_t)IN0P * NFEAT * 2);
    unsigned short* w128[10];
    for (int i = 0; i < 10; ++i)
        w128[i] = (unsigned short*)ALLOC((size_t)NFEAT * NFEAT * 2);
    // w128: 0=rg0_c1_w0 1=rg0_c1_w1 2..5=rg1 6..9=rg2
    unsigned short* gcw0b = (unsigned short*)ALLOC((size_t)NFEAT * 16 * 2);
    unsigned short* gcw1b = (unsigned short*)ALLOC((size_t)NFEAT * 16 * 2);

    float*          P   = (float*)ALLOC((size_t)NVERT * NFEAT * 4);
    float*          S   = (float*)ALLOC((size_t)NVERT * NFEAT * 4);
    float*          T   = (float*)ALLOC((size_t)NVERT * NFEAT * 4);
    unsigned short* Tb  = (unsigned short*)ALLOC((size_t)NVERT * NFEAT * 2);
    float*          U   = (float*)ALLOC((size_t)NVERT * NFEAT * 4);
    float*          G1  = (float*)ALLOC((size_t)NVERT * NFEAT * 4);
    unsigned short* G1b = (unsigned short*)ALLOC((size_t)NVERT * NFEAT * 2);
    float*          G2  = (float*)ALLOC((size_t)NVERT * NFEAT * 4);
    unsigned short* G2b = (unsigned short*)ALLOC((size_t)NVERT * NFEAT * 2);
    float*          X   = (float*)ALLOC((size_t)NVERT * NFEAT * 4);
    unsigned short* Xb  = (unsigned short*)ALLOC((size_t)NVERT * NFEAT * 2);
    float*          D   = (float*)ALLOC((size_t)NVERT * 16 * 4);
    unsigned short* Db  = (unsigned short*)ALLOC((size_t)NVERT * 16 * 2);
    float*          DU  = (float*)ALLOC((size_t)NVERT * 16 * 4);

    // ---- adj -> bf16 (exact: entries are 0/1); bf16 copy fits in 192MB L2
    cvt_bf16_vec_kernel<<<4096, 256, 0, stream>>>(adj, adjb,
                                                  (size_t)NVERT * NVERT);
    // ---- weights -> bf16 (zero-padded)
    auto CVT = [&](const float* s, unsigned short* d, int K, int N, int Kp, int Np) {
        int tot = Kp * Np;
        int blocks = (tot + 255) / 256;
        if (blocks > 1024) blocks = 1024;
        cvt_pad_kernel<<<blocks, 256, 0, stream>>>(s, d, K, N, Kp, Np);
    };
    CVT(lin_w, linb, ALIGN_CH, NFEAT, ALIGN_CH, NFEAT);
    CVT(W[0], wpad0, IN0, NFEAT, IN0P, NFEAT);
    CVT(W[1], wpad1, IN0, NFEAT, IN0P, NFEAT);
    CVT(W[4], wproj, IN0, NFEAT, IN0P, NFEAT);
    CVT(W[2], w128[0], NFEAT, NFEAT, NFEAT, NFEAT);
    CVT(W[3], w128[1], NFEAT, NFEAT, NFEAT, NFEAT);
    for (int i = 0; i < 4; ++i) CVT(W[5 + i], w128[2 + i], NFEAT, NFEAT, NFEAT, NFEAT);
    for (int i = 0; i < 4; ++i) CVT(W[9 + i], w128[6 + i], NFEAT, NFEAT, NFEAT, NFEAT);
    CVT(W[13], gcw0b, NFEAT, 3, NFEAT, 16);
    CVT(W[14], gcw1b, NFEAT, 3, NFEAT, 16);

    // ---- vertex_align gather (features are L2-resident)
    vertex_align_kernel<<<NVERT, 256, 0, stream>>>(f0, f1, f2, f3, pos, alignedb);

    auto GEMM = [&](const unsigned short* A, const unsigned short* B, float* Cf,
                    unsigned short* Cb, const float* Aux, int M, int N, int K,
                    int relu) {
        dim3 grid(M / 128, (N + 127) / 128);
        if (N % 128 == 0)
            gemm_bf16_wmma<8><<<grid, 256, 0, stream>>>(A, B, Cf, Cb, Aux, M, N, K, relu);
        else
            gemm_bf16_wmma<1><<<grid, 256, 0, stream>>>(A, B, Cf, Cb, Aux, M, N, K, relu);
    };

    // ---- projected = aligned @ lin_w ; x0 = concat(vfeat, pos, projected)
    GEMM(alignedb, linb, P, nullptr, nullptr, NVERT, NFEAT, ALIGN_CH, 0);
    concat_kernel<<<NVERT, 256, 0, stream>>>(vfeat, pos, P, x0b);

    size_t nX = (size_t)NVERT * NFEAT;
    int addBlocks = (int)((nX + 255) / 256);

    // ---- rg0 (with projection skip)
    GEMM(x0b, wproj, S, nullptr, nullptr, NVERT, NFEAT, IN0P, 0);          // skip
    GEMM(x0b, wpad1, T, Tb, nullptr, NVERT, NFEAT, IN0P, 0);               // x@w1
    GEMM(adjb, Tb, U, nullptr, nullptr, NVERT, NFEAT, NVERT, 0);           // adj@
    GEMM(x0b, wpad0, G1, G1b, U, NVERT, NFEAT, IN0P, 1);                   // relu(+)
    GEMM(G1b, w128[1], T, Tb, nullptr, NVERT, NFEAT, NFEAT, 0);
    GEMM(adjb, Tb, U, nullptr, nullptr, NVERT, NFEAT, NVERT, 0);
    GEMM(G1b, w128[0], G2, G2b, U, NVERT, NFEAT, NFEAT, 1);
    add_dual_kernel<<<addBlocks, 256, 0, stream>>>(S, G2, X, Xb, nX);

    // ---- rg1, rg2 (identity skip, in-place accumulate into X)
    for (int L = 0; L < 2; ++L) {
        unsigned short* c0w0 = w128[2 + L * 4 + 0];
        unsigned short* c0w1 = w128[2 + L * 4 + 1];
        unsigned short* c1w0 = w128[2 + L * 4 + 2];
        unsigned short* c1w1 = w128[2 + L * 4 + 3];
        GEMM(Xb, c0w1, T, Tb, nullptr, NVERT, NFEAT, NFEAT, 0);
        GEMM(adjb, Tb, U, nullptr, nullptr, NVERT, NFEAT, NVERT, 0);
        GEMM(Xb, c0w0, G1, G1b, U, NVERT, NFEAT, NFEAT, 1);
        GEMM(G1b, c1w1, T, Tb, nullptr, NVERT, NFEAT, NFEAT, 0);
        GEMM(adjb, Tb, U, nullptr, nullptr, NVERT, NFEAT, NVERT, 0);
        GEMM(G1b, c1w0, G2, G2b, U, NVERT, NFEAT, NFEAT, 1);
        add_dual_kernel<<<addBlocks, 256, 0, stream>>>(X, G2, X, Xb, nX);
    }

    // ---- gc head (N padded 3 -> 16), delta = tanh(relu(...))
    GEMM(Xb, gcw1b, D, Db, nullptr, NVERT, 16, NFEAT, 0);
    GEMM(adjb, Db, DU, nullptr, nullptr, NVERT, 16, NVERT, 0);
    GEMM(Xb, gcw0b, D, nullptr, DU, NVERT, 16, NFEAT, 1);

    final_kernel<<<addBlocks, 256, 0, stream>>>(pos, D, X, (float*)d_out);
}
